// FrFDConv_53798760350303
// MI455X (gfx1250) — matrix operand order repaired and checked by reading, not compiled, not run on hardware
//
#include <hip/hip_runtime.h>
#include <hip/hip_bf16.h>
#include <math.h>

// ---------------------------------------------------------------------------
// FrFDConv on gfx1250: pooled gate -> per-batch 16x16x3x3 f16 kernels ->
// implicit-GEMM conv with V_WMMA_F32_16X16X32_F16 (f32 accumulate).
// K-dim ordering = (tap = ky*3+kx, ic), padded 144 -> 160.
// ---------------------------------------------------------------------------

typedef __attribute__((ext_vector_type(16))) _Float16 v16h;
typedef __attribute__((ext_vector_type(8)))  float    v8f;
typedef __attribute__((ext_vector_type(4)))  unsigned int v4u;
typedef __attribute__((ext_vector_type(4)))  float    v4f;

static __device__ __forceinline__ v16h make_v16h(v4u lo, v4u hi) {
    union { v4u u[2]; v16h h; } t;
    t.u[0] = lo; t.u[1] = hi;
    return t.h;
}

#define B_     16
#define C_     16
#define HW_    512
#define OW_    510
#define G_     64
#define KPAD   160   // 144 real K (9 taps * 16 ic) padded to 5 chunks of 32
#define PLANE  (510 * 510)

// ---------------------------------------------------------------------------
// Kernel 1: global average pool. One block per (b,c); 256 threads; float4 loads.
// ---------------------------------------------------------------------------
__global__ __launch_bounds__(256)
void frfd_pool(const float* __restrict__ x, float* __restrict__ pooled) {
    const int bc  = blockIdx.x;          // 0..255
    const int tid = threadIdx.x;
    const v4f* p4 = (const v4f*)(x + (size_t)bc * (HW_ * HW_));
    float s = 0.f;
    for (int k = tid; k < (HW_ * HW_) / 4; k += 256) {
        v4f v = p4[k];
        s += v[0] + v[1] + v[2] + v[3];
    }
    __shared__ float red[256];
    red[tid] = s;
    __syncthreads();
    for (int off = 128; off > 0; off >>= 1) {
        if (tid < off) red[tid] += red[tid + off];
        __syncthreads();
    }
    if (tid == 0) pooled[bc] = red[0] * (1.0f / (HW_ * HW_));
}

// ---------------------------------------------------------------------------
// Kernel 2: gate + dynamic weight build. One block per batch.
// Produces wf16[b][oc][tap*16 + ic] (f16), zero-padded for K in [144,160).
// Radial group map: gid = floor(sqrt(gi^2+gj^2) * 128), valid iff idx < 0.5.
// ---------------------------------------------------------------------------
__global__ __launch_bounds__(256)
void frfd_gate_weights(const float* __restrict__ pooled,
                       const float* __restrict__ spectral,
                       const float* __restrict__ attn_w,
                       const float* __restrict__ attn_b,
                       _Float16* __restrict__ wf16) {
    const int b   = blockIdx.x;
    const int tid = threadIdx.x;
    __shared__ float pl[C_];
    __shared__ float pi[G_];
    if (tid < C_) pl[tid] = pooled[b * C_ + tid];
    __syncthreads();
    if (tid < G_) {
        float s = attn_b[tid];
        #pragma unroll
        for (int c = 0; c < C_; ++c) s += pl[c] * attn_w[tid * C_ + c];
        pi[tid] = 1.0f / (1.0f + __expf(-s));
    }
    __syncthreads();

    for (int i = tid; i < C_ * KPAD; i += 256) {  // 2560 entries
        const int oc = i / KPAD;
        const int k  = i % KPAD;
        _Float16 val = (_Float16)0.0f;
        if (k < 144) {
            const int tap = k >> 4;      // 0..8
            const int ic  = k & 15;
            const int ky  = tap / 3, kx = tap % 3;
            const int ir  = ic * 3 + ky; // spectral row  (0..47)
            const int jc  = oc * 3 + kx; // spectral col  (0..47)
            const float gi = -0.5f + (float)ir * (1.0f / 47.0f);
            const float gj = -0.5f + (float)jc * (1.0f / 47.0f);
            const float idx = sqrtf(gi * gi + gj * gj);
            const int gid = (int)floorf(idx * 128.0f);   // thresholds step = 1/128
            const float gate = (idx < 0.5f && gid >= 0 && gid < G_) ? pi[gid] : 0.0f;
            val = (_Float16)(spectral[ir * 48 + jc] * gate);
        }
        wf16[(size_t)b * (C_ * KPAD) + i] = val;
    }
}

// ---------------------------------------------------------------------------
// Kernel 3: implicit-GEMM conv with WMMA.
// grid = (32 x-tiles, 16 y-tiles, 16 batches); block = 256 (8 waves).
// Block tile: 16 px wide x 32 px tall; wave w handles 4 output rows.
// LDS patch: 34 rows x 18 cols x 16 ic (f16), entry stride 24 halves (48 B)
// so the per-lane ds_load_b128 pairs are 16B-aligned and bank-conflict-free.
// ---------------------------------------------------------------------------
#define PAT_R 34
#define PAT_C 18
#define ICST  24   // halves per (row,col) entry (16 data + 8 pad)

__global__ __launch_bounds__(256)
void frfd_conv_wmma(const float* __restrict__ x,
                    const _Float16* __restrict__ wf16,
                    float* __restrict__ out) {
    const int x0 = blockIdx.x * 16;
    const int y0 = blockIdx.y * 32;
    const int b  = blockIdx.z;
    const int tid  = threadIdx.x;
    const int lane = tid & 31;
    const int w    = tid >> 5;        // wave id 0..7
    const int sel  = (lane >= 16) ? 1 : 0;
    const int coln = lane & 15;       // pixel column within tile / matrix row M

    __shared__ _Float16 lx[PAT_R * PAT_C * ICST];   // ~29.4 KB

    // ---- stage input patch (f32 -> f16), clamped at image edges ----------
    for (int i = tid; i < PAT_R * PAT_C * C_; i += 256) {
        const int col = i % PAT_C;
        const int t1  = i / PAT_C;
        const int row = t1 % PAT_R;
        const int ic  = t1 / PAT_R;
        const int gy = min(y0 + row, HW_ - 1);
        const int gx = min(x0 + col, HW_ - 1);
        const float v = x[(((size_t)b * C_ + ic) * HW_ + gy) * HW_ + gx];
        lx[(row * PAT_C + col) * ICST + ic] = (_Float16)v;
    }

    // ---- preload A (per-batch weights) into registers, all 5 K-chunks ----
    // A lane layout (16-bit A 16x32): lanes<16: h0..15=K(kc+0..15)
    //                                 lanes>=16: h0..15=K(kc+16..31)
    const _Float16* wrow = wf16 + ((size_t)b * C_ + coln) * KPAD;
    v16h A[5];
    #pragma unroll
    for (int c = 0; c < 5; ++c) {
        const v4u lo = *(const v4u*)(wrow + c * 32 + sel * 16);
        const v4u hi = *(const v4u*)(wrow + c * 32 + sel * 16 + 8);
        A[c] = make_v16h(lo, hi);
    }

    __syncthreads();

    // ---- compute: 4 output rows per wave, 5 WMMAs each -------------------
    #pragma unroll
    for (int rr = 0; rr < 4; ++rr) {
        const int rloc = w * 4 + rr;          // local output row 0..31
        v8f acc = {};
        #pragma unroll
        for (int c = 0; c < 5; ++c) {
            // B lane layout (32x16 f16): lanes<16: h=K(kc+0..15) -> tap 2c, ic 0..15
            //                            lanes>=16: h=K(kc+16..31) -> tap 2c+1
            int tap = 2 * c + sel;
            if (tap > 8) tap = 8;             // tap 9 pairs with zero A columns
            const int ky = tap / 3, kx = tap % 3;
            const int entry = (rloc + ky) * PAT_C + (coln + kx);
            const v4u* q = (const v4u*)&lx[entry * ICST];
            const v16h Bm = make_v16h(q[0], q[1]);
            acc = __builtin_amdgcn_wmma_f32_16x16x32_f16(
                false, A[c], false, Bm, (short)0, acc, false, false);
        }
        // ---- store: VGPR v -> M = v + sel*8, N = coln ---------------------
        const int y  = y0 + rloc;
        const int xg = x0 + coln;
        if (y < OW_ && xg < OW_) {
            float* op = out + (((size_t)b * C_ + sel * 8) * OW_ + y) * OW_ + xg;
            #pragma unroll
            for (int v = 0; v < 8; ++v)
                op[(size_t)v * PLANE] = acc[v];
        }
    }
}

// ---------------------------------------------------------------------------
// Launch
// ---------------------------------------------------------------------------
extern "C" void kernel_launch(void* const* d_in, const int* in_sizes, int n_in,
                              void* d_out, int out_size, void* d_ws, size_t ws_size,
                              hipStream_t stream) {
    const float* x        = (const float*)d_in[0];   // [16,16,512,512]
    const float* spectral = (const float*)d_in[1];   // [48,48]
    const float* attn_w   = (const float*)d_in[2];   // [64,16]
    const float* attn_b   = (const float*)d_in[3];   // [64]
    float* out = (float*)d_out;                      // [16,16,510,510]

    float*    pooled = (float*)d_ws;                            // 256 floats
    _Float16* wf16   = (_Float16*)((char*)d_ws + 1024);         // 16*16*160 halves

    frfd_pool<<<256, 256, 0, stream>>>(x, pooled);
    frfd_gate_weights<<<B_, 256, 0, stream>>>(pooled, spectral, attn_w, attn_b, wf16);

    dim3 grid(32, 16, B_);   // x-tiles, y-tiles, batch
    frfd_conv_wmma<<<grid, 256, 0, stream>>>(x, wf16, out);
}
